// SequentialMLP_30279519436987
// MI455X (gfx1250) — matrix operand order repaired and checked by reading, not compiled, not run on hardware
//
#include <hip/hip_runtime.h>
#include <hip/hip_bf16.h>

// MoE grouped MLP (SwiGLU) for MI455X / gfx1250 (wave32, WMMA bf16).
// Phase 0: one-time layout pass into d_ws:
//   Xb  = bf16(X)                     [E][TPE][HID]   (row-major, K-contig)
//   Wgt = bf16(gate_w^T)              [E][INTER][HID] ([n][k], K-contig)
//   Wut = bf16(up_w^T)                [E][INTER][HID]
//   Wdt = bf16(down_w^T)              [E][HID][INTER]
//   Hb  = bf16 SwiGLU intermediate    [E][TPE][INTER]
// Phase 1: fused gate+up GEMM + SwiGLU (async-LDS double-buffered, bf16 WMMA)
// Phase 2: down GEMM -> fp32 out.
// Compute-bound (AI ~740 FLOP/B @ 23.3 TB/s) => v_wmma_f32_16x16x32_bf16.

#define NEXP   8
#define HID    2048
#define INTER  5504
#define TPE    2048
#define TOKENS 16384

typedef __attribute__((ext_vector_type(16))) __bf16 bf16x16;
typedef __attribute__((ext_vector_type(8)))  __bf16 bf16x8;
typedef __attribute__((ext_vector_type(8)))  float  v8f;

static __device__ __forceinline__ bf16x16 cat8(bf16x8 lo, bf16x8 hi) {
  return __builtin_shufflevector(lo, hi, 0,1,2,3,4,5,6,7,8,9,10,11,12,13,14,15);
}

static __device__ __forceinline__ v8f wmma_bf16(bf16x16 a, bf16x16 b, v8f c) {
  return __builtin_amdgcn_wmma_f32_16x16x32_bf16(false, a, false, b, (short)0, c, false, false);
}

// A fragment (16x32): lanes 0-15 = rows M; lane half lh selects K packing
// [lh*8 .. +7] and [16+lh*8 .. +7] -> two contiguous 16B LDS reads.
static __device__ __forceinline__ bf16x16 load_a_frag(const __bf16* rowPtr, int lh) {
  bf16x8 lo = *(const bf16x8*)(rowPtr + lh * 8);
  bf16x8 hi = *(const bf16x8*)(rowPtr + 16 + lh * 8);
  return cat8(lo, hi);
}

// B fragment (32x16) from [n][k] LDS tile: lane = col N; lane half lh selects K
// [lh*8 .. +7] and [16+lh*8 .. +7] (same as A) -> two 16B reads.
static __device__ __forceinline__ bf16x16 load_b_frag(const __bf16* colPtr, int lh) {
  const __bf16* p = colPtr + lh * 8;
  bf16x8 lo = *(const bf16x8*)(p);
  bf16x8 hi = *(const bf16x8*)(p + 16);
  return cat8(lo, hi);
}

// Low 32 bits of a flat shared-space address are the LDS byte offset.
static __device__ __forceinline__ unsigned lds_addr(const void* p) {
  return (unsigned)(size_t)p;
}

// CDNA5 async global->LDS copy, 16B per lane. Tracked by ASYNCcnt.
static __device__ __forceinline__ void async_b128(const __bf16* g, unsigned lds) {
  asm volatile("global_load_async_to_lds_b128 %0, %1, off"
               :: "v"(lds), "v"(g) : "memory");
}
static __device__ __forceinline__ void wait_async0() {
  asm volatile("s_wait_asynccnt 0x0" ::: "memory");
}

// ---------------- Phase 0a: elementwise fp32 -> bf16 ----------------
__global__ __launch_bounds__(256)
void cvt_bf16_kernel(const float* __restrict__ src, __bf16* __restrict__ dst) {
  size_t i = ((size_t)blockIdx.x * 256 + threadIdx.x) * 8;
  float4 a = *(const float4*)(src + i);
  float4 b = *(const float4*)(src + i + 4);
  __attribute__((aligned(16))) __bf16 t[8];
  t[0] = (__bf16)a.x; t[1] = (__bf16)a.y; t[2] = (__bf16)a.z; t[3] = (__bf16)a.w;
  t[4] = (__bf16)b.x; t[5] = (__bf16)b.y; t[6] = (__bf16)b.z; t[7] = (__bf16)b.w;
  *(uint4*)(dst + i) = *(const uint4*)t;
}

// ---------- Phase 0b: transpose-convert fp32 [R][C] -> bf16 [C][R] ----------
// grid (C/64, R/64, NEXP), block 256. 64x64 tile via LDS.
__global__ __launch_bounds__(256)
void transpose_cvt_kernel(const float* __restrict__ src, __bf16* __restrict__ dst,
                          int R, int C) {
  __shared__ __attribute__((aligned(16))) __bf16 st[64][72]; // 144B row pitch (16B mult)
  const int e  = blockIdx.z;
  const int r0 = blockIdx.y * 64;
  const int c0 = blockIdx.x * 64;
  src += (size_t)e * R * C;
  dst += (size_t)e * R * C;
  const int tid = threadIdx.x;
  #pragma unroll
  for (int i = 0; i < 4; ++i) {
    int id = tid + i * 256;          // 0..1023
    int r  = id >> 4;                // 0..63
    int c  = (id & 15) * 4;          // 0..60
    float4 v = *(const float4*)(src + (size_t)(r0 + r) * C + c0 + c);
    st[c + 0][r] = (__bf16)v.x;
    st[c + 1][r] = (__bf16)v.y;
    st[c + 2][r] = (__bf16)v.z;
    st[c + 3][r] = (__bf16)v.w;
  }
  __syncthreads();
  #pragma unroll
  for (int i = 0; i < 2; ++i) {
    int id = tid + i * 256;          // 0..511
    int c  = id >> 3;                // 0..63
    int r  = (id & 7) * 8;           // 0..56
    uint4 v = *(const uint4*)&st[c][r];
    *(uint4*)(dst + (size_t)(c0 + c) * R + r0 + r) = v;
  }
}

// Issue one 128x32 bf16 tile fill (8KB) via async copies; K-contiguous source
// rows of length `stride`. 2 x b128 per thread (256 threads).
static __device__ __forceinline__ void stage_tile(const __bf16* gbase, unsigned ldsBase,
                                                  int k0, int tid, int stride) {
  #pragma unroll
  for (int i = 0; i < 2; ++i) {
    int c   = tid + i * 256;         // 0..511
    int row = c >> 2;                // 0..127
    int kc  = (c & 3) * 8;           // 0,8,16,24
    async_b128(gbase + (size_t)row * stride + k0 + kc,
               ldsBase + (unsigned)(row * 32 + kc) * 2);
  }
}

// ---------------- Phase 1: fused gate/up GEMM + SwiGLU ----------------
// grid (INTER/128, TPE/128, NEXP), block 256 (8 waves, 4x2; wave tile 32x64)
__global__ __launch_bounds__(256)
void moe_gateup_swiglu(const __bf16* __restrict__ Xb,
                       const __bf16* __restrict__ Wgt,
                       const __bf16* __restrict__ Wut,
                       __bf16* __restrict__ Hb) {
  __shared__ __attribute__((aligned(16))) __bf16 sA[2][128 * 32];
  __shared__ __attribute__((aligned(16))) __bf16 sG[2][128 * 32];
  __shared__ __attribute__((aligned(16))) __bf16 sU[2][128 * 32];

  const int e  = blockIdx.z;
  const int m0 = blockIdx.y * 128;
  const int n0 = blockIdx.x * 128;

  const __bf16* x  = Xb  + (size_t)e * TPE * HID + (size_t)m0 * HID;
  const __bf16* wg = Wgt + (size_t)e * INTER * HID + (size_t)n0 * HID;
  const __bf16* wu = Wut + (size_t)e * INTER * HID + (size_t)n0 * HID;
  __bf16*       ho = Hb  + (size_t)e * TPE * INTER + (size_t)m0 * INTER + n0;

  const int tid = threadIdx.x, lane = tid & 31, wv = tid >> 5;
  const int wm = (wv >> 1) * 32, wn = (wv & 1) * 64;
  const int lh = lane >> 4, ll = lane & 15;

  const unsigned aB[2] = { lds_addr(&sA[0][0]), lds_addr(&sA[1][0]) };
  const unsigned gB[2] = { lds_addr(&sG[0][0]), lds_addr(&sG[1][0]) };
  const unsigned uB[2] = { lds_addr(&sU[0][0]), lds_addr(&sU[1][0]) };

  v8f accG[2][4] = {}; v8f accU[2][4] = {};

  // prologue: fill buffer 0
  stage_tile(x,  aB[0], 0, tid, HID);
  stage_tile(wg, gB[0], 0, tid, HID);
  stage_tile(wu, uB[0], 0, tid, HID);
  wait_async0();
  __syncthreads();

  for (int k0 = 0; k0 < HID; k0 += 32) {
    const int cur = (k0 >> 5) & 1, nxt = cur ^ 1;
    if (k0 + 32 < HID) {               // prefetch next K-slab into other buffer
      stage_tile(x,  aB[nxt], k0 + 32, tid, HID);
      stage_tile(wg, gB[nxt], k0 + 32, tid, HID);
      stage_tile(wu, uB[nxt], k0 + 32, tid, HID);
    }
    bf16x16 aF[2];
    #pragma unroll
    for (int mt = 0; mt < 2; ++mt)
      aF[mt] = load_a_frag(&sA[cur][(wm + mt * 16 + ll) * 32], lh);
    #pragma unroll
    for (int nt = 0; nt < 4; ++nt) {
      bf16x16 gF = load_b_frag(&sG[cur][(wn + nt * 16 + ll) * 32], lh);
      bf16x16 uF = load_b_frag(&sU[cur][(wn + nt * 16 + ll) * 32], lh);
      #pragma unroll
      for (int mt = 0; mt < 2; ++mt) {
        accG[mt][nt] = wmma_bf16(aF[mt], gF, accG[mt][nt]);
        accU[mt][nt] = wmma_bf16(aF[mt], uF, accU[mt][nt]);
      }
    }
    wait_async0();                     // our own async fills landed
    __syncthreads();                   // everyone done reading cur / filling nxt
  }

  // SwiGLU epilogue: h = silu(g) * u, bf16 store
  #pragma unroll
  for (int mt = 0; mt < 2; ++mt)
    #pragma unroll
    for (int nt = 0; nt < 4; ++nt) {
      v8f g = accG[mt][nt], u = accU[mt][nt];
      #pragma unroll
      for (int r = 0; r < 8; ++r) {
        float gv = g[r];
        float hv = (gv / (1.0f + __expf(-gv))) * u[r];
        int row = wm + mt * 16 + lh * 8 + r;   // C layout: M = 8*half + r
        int col = wn + nt * 16 + ll;           // N = lane&15
        ho[(size_t)row * INTER + col] = (__bf16)hv;
      }
    }
}

// ---------------- Phase 2: down projection ----------------
// grid (HID/128, TPE/128, NEXP), block 256
__global__ __launch_bounds__(256)
void moe_down(const __bf16* __restrict__ Hb,
              const __bf16* __restrict__ Wdt,
              float* __restrict__ Out) {
  __shared__ __attribute__((aligned(16))) __bf16 sA[2][128 * 32];
  __shared__ __attribute__((aligned(16))) __bf16 sB[2][128 * 32];

  const int e  = blockIdx.z;
  const int m0 = blockIdx.y * 128;
  const int n0 = blockIdx.x * 128;

  const __bf16* hx = Hb  + (size_t)e * TPE * INTER + (size_t)m0 * INTER;
  const __bf16* wd = Wdt + (size_t)e * HID * INTER + (size_t)n0 * INTER;
  float*        oo = Out + (size_t)e * TPE * HID + (size_t)m0 * HID + n0;

  const int tid = threadIdx.x, lane = tid & 31, wv = tid >> 5;
  const int wm = (wv >> 1) * 32, wn = (wv & 1) * 64;
  const int lh = lane >> 4, ll = lane & 15;

  const unsigned aB[2] = { lds_addr(&sA[0][0]), lds_addr(&sA[1][0]) };
  const unsigned bB[2] = { lds_addr(&sB[0][0]), lds_addr(&sB[1][0]) };

  v8f acc[2][4] = {};

  stage_tile(hx, aB[0], 0, tid, INTER);
  stage_tile(wd, bB[0], 0, tid, INTER);
  wait_async0();
  __syncthreads();

  for (int k0 = 0; k0 < INTER; k0 += 32) {
    const int cur = (k0 >> 5) & 1, nxt = cur ^ 1;
    if (k0 + 32 < INTER) {
      stage_tile(hx, aB[nxt], k0 + 32, tid, INTER);
      stage_tile(wd, bB[nxt], k0 + 32, tid, INTER);
    }
    bf16x16 aF[2];
    #pragma unroll
    for (int mt = 0; mt < 2; ++mt)
      aF[mt] = load_a_frag(&sA[cur][(wm + mt * 16 + ll) * 32], lh);
    #pragma unroll
    for (int nt = 0; nt < 4; ++nt) {
      bf16x16 bF = load_b_frag(&sB[cur][(wn + nt * 16 + ll) * 32], lh);
      #pragma unroll
      for (int mt = 0; mt < 2; ++mt)
        acc[mt][nt] = wmma_bf16(aF[mt], bF, acc[mt][nt]);
    }
    wait_async0();
    __syncthreads();
  }

  #pragma unroll
  for (int mt = 0; mt < 2; ++mt)
    #pragma unroll
    for (int nt = 0; nt < 4; ++nt) {
      v8f c = acc[mt][nt];
      #pragma unroll
      for (int r = 0; r < 8; ++r) {
        int row = wm + mt * 16 + lh * 8 + r;
        int col = wn + nt * 16 + ll;
        oo[(size_t)row * HID + col] = c[r];
      }
    }
}

extern "C" void kernel_launch(void* const* d_in, const int* in_sizes, int n_in,
                              void* d_out, int out_size, void* d_ws, size_t ws_size,
                              hipStream_t stream) {
  const float* X  = (const float*)d_in[0];  // [TOKENS, HID]
  const float* Wg = (const float*)d_in[1];  // [E, HID, INTER]
  const float* Wu = (const float*)d_in[2];  // [E, HID, INTER]
  const float* Wd = (const float*)d_in[3];  // [E, INTER, HID]
  // d_in[4] tokens_per_expert: uniform per reference -> static tiling.

  const size_t nX = (size_t)TOKENS * HID;             // 33,554,432
  const size_t nW = (size_t)NEXP * HID * INTER;       // 90,177,536
  __bf16* Xb  = (__bf16*)d_ws;                        // bf16 activations
  __bf16* Wgt = Xb  + nX;                             // [E][INTER][HID]
  __bf16* Wut = Wgt + nW;                             // [E][INTER][HID]
  __bf16* Wdt = Wut + nW;                             // [E][HID][INTER]
  __bf16* Hb  = Wdt + nW;                             // [E][TPE][INTER]
  float*  Out = (float*)d_out;
  // total ws need: (nX + 4*nW) * 2 bytes ~= 752 MiB

  dim3 blk(256);
  // Phase 0
  cvt_bf16_kernel<<<dim3((unsigned)(nX / 8 / 256)), blk, 0, stream>>>(X, Xb);
  transpose_cvt_kernel<<<dim3(INTER / 64, HID / 64, NEXP), blk, 0, stream>>>(Wg, Wgt, HID, INTER);
  transpose_cvt_kernel<<<dim3(INTER / 64, HID / 64, NEXP), blk, 0, stream>>>(Wu, Wut, HID, INTER);
  transpose_cvt_kernel<<<dim3(HID / 64, INTER / 64, NEXP), blk, 0, stream>>>(Wd, Wdt, INTER, HID);
  // Phase 1 + 2
  moe_gateup_swiglu<<<dim3(INTER / 128, TPE / 128, NEXP), blk, 0, stream>>>(Xb, Wgt, Wut, Hb);
  moe_down<<<dim3(HID / 128, TPE / 128, NEXP), blk, 0, stream>>>(Hb, Wdt, Out);
}
